// LongformerSelfAttention_73804718014557
// MI455X (gfx1250) — matrix operand order repaired and checked by reading, not compiled, not run on hardware
//
#include <hip/hip_runtime.h>

typedef _Float16 h16;
typedef __attribute__((ext_vector_type(16))) _Float16 v16h;
typedef __attribute__((ext_vector_type(8)))  _Float16 v8h;
typedef __attribute__((ext_vector_type(8)))  float    v8f;

#define NB 2
#define NS 4096
#define ND 768
#define NH 12
#define DH 64
#define NW 256
#define BS (NB*NS)          /* 8192 rows */
#define NEGF (-1.0e30f)

static __device__ __forceinline__ v16h cat8(v8h a, v8h b) {
  v16h r;
#pragma unroll
  for (int i = 0; i < 8; i++) { r[i] = a[i]; r[8 + i] = b[i]; }
  return r;
}

static __device__ __forceinline__ v8f wmma_f16(v16h a, v16h b, v8f c) {
  // D = A(16x32 f16) * B(32x16 f16) + C(16x16 f32)
  return __builtin_amdgcn_wmma_f32_16x16x32_f16(false, a, false, b, (short)0, c, false, false);
}

// ---------------- conversion kernels ----------------
__global__ void cvt_x(const float* __restrict__ X, h16* __restrict__ Xh, int n) {
  int i = blockIdx.x * blockDim.x + threadIdx.x;
  if (i < n) Xh[i] = (h16)X[i];
}

// WT[mat][colOut][feat] = W[feat][colOut]  (transposed, f16)
__global__ void cvt_wT(const float* __restrict__ Wq, const float* __restrict__ Wk,
                       const float* __restrict__ Wv, h16* __restrict__ WT) {
  int i = blockIdx.x * blockDim.x + threadIdx.x;
  if (i >= 3 * ND * ND) return;
  int mat = i / (ND * ND);
  int r = (i / ND) % ND;   // output column
  int c = i % ND;          // input feature
  const float* Wsrc = mat == 0 ? Wq : (mat == 1 ? Wk : Wv);
  WT[i] = (h16)Wsrc[c * ND + r];
}

// ---------------- QKV projection: one wave = one 32x64 tile ----------------
// Per K-step: 2 A-fragments + 4 B-fragments feed 8 WMMAs (A reused x4, B x2).
__global__ __launch_bounds__(32) void qkv_gemm(
    const h16* __restrict__ Xh, const h16* __restrict__ WT,
    const float* __restrict__ bq, const float* __restrict__ bk, const float* __restrict__ bv,
    h16* __restrict__ Qh, h16* __restrict__ Kh, h16* __restrict__ VTh) {
  const int lane = threadIdx.x & 31;
  const int mat  = blockIdx.y;
  const int mt   = blockIdx.x / (ND / 64);   // 256 row-tiles of 32
  const int nb   = blockIdx.x % (ND / 64);   // 12 col-tiles of 64 == head index
  const int l16  = lane & 15;
  const int hi   = lane >> 4;
  const int a0 = hi ? 16 : 0, a1 = hi ? 24 : 8, b0 = hi ? 16 : 0;

  const h16* xr0 = Xh + (size_t)(mt * 32 + l16) * ND;        // A rows, M-tile 0
  const h16* xr1 = xr0 + (size_t)16 * ND;                    // A rows, M-tile 1
  const h16* wtb = WT + (size_t)mat * ND * ND + (size_t)(nb * 64 + l16) * ND;

  v8f acc[8] = {};   // acc[mi*4 + j]
#pragma unroll 2
  for (int ks = 0; ks < ND / 32; ks++) {
    int kb = ks * 32;
    v16h A0 = cat8(*(const v8h*)(xr0 + kb + a0), *(const v8h*)(xr0 + kb + a1));
    v16h A1 = cat8(*(const v8h*)(xr1 + kb + a0), *(const v8h*)(xr1 + kb + a1));
#pragma unroll
    for (int j = 0; j < 4; j++) {
      v16h Bf = *(const v16h*)(wtb + (size_t)(j * 16) * ND + kb + b0);
      acc[j]     = wmma_f16(A0, Bf, acc[j]);
      acc[4 + j] = wmma_f16(A1, Bf, acc[4 + j]);
    }
  }

  const float* bias = mat == 0 ? bq : (mat == 1 ? bk : bv);
  const float scale = (mat == 0) ? 0.125f : 1.0f;            // 1/sqrt(64) on Q
#pragma unroll
  for (int j = 0; j < 4; j++) {
    const int dh = j * 16 + l16;                             // col within head
    const float bc = bias[nb * 64 + dh];
#pragma unroll
    for (int mi = 0; mi < 2; mi++) {
#pragma unroll
      for (int r = 0; r < 8; r++) {
        int sr = mt * 32 + mi * 16 + r + (hi ? 8 : 0);       // global row b*S+s
        int bi = sr >> 12;
        int s  = sr & (NS - 1);
        h16 yh = (h16)((acc[mi * 4 + j][r] + bc) * scale);
        if (mat == 0)
          Qh[(((size_t)(bi * NH + nb) * NS + s) << 6) + dh] = yh;
        else if (mat == 1)
          Kh[(((size_t)(bi * NH + nb) * NS + s) << 6) + dh] = yh;
        else
          VTh[((size_t)(bi * NH + nb) * DH + dh) * NS + s] = yh;
      }
    }
  }
}

// ---------------- one 32-key attention step ----------------
static __device__ __forceinline__ void attn_step(
    int k0, bool edge, int q0, int hi, int l16,
    const h16* __restrict__ Kbh, const h16* __restrict__ VTbh,
    const float* __restrict__ amrow,
    v16h Aq0, v16h Aq1, h16 (*__restrict__ slab)[32],
    int a0, int a1, int b0off,
    float* __restrict__ m_run, float* __restrict__ l_run, v8f* __restrict__ o) {
  // ---- scores: two 16-key tiles ----
  v8f s[2];
#pragma unroll
  for (int n = 0; n < 2; n++) {
    int kt = k0 + n * 16;
    int kc = kt < 0 ? 0 : (kt > NS - 16 ? NS - 16 : kt);     // 16-aligned group clamp
    const h16* krow = Kbh + (size_t)(kc + l16) * DH;
    v16h B0 = *(const v16h*)(krow + b0off);
    v16h B1 = *(const v16h*)(krow + 32 + b0off);
    v8f sa = {};
    sa = wmma_f16(Aq0, B0, sa);
    sa = wmma_f16(Aq1, B1, sa);

    int kg = kt + l16;
    if (edge) {                         // band + range masking (boundary steps only)
      bool okc = (kg >= 0) && (kg < NS);
      float fm = (okc && amrow[kg] != 0.f) ? NEGF : 0.f;
#pragma unroll
      for (int r = 0; r < 8; r++) {
        int qg = q0 + r + (hi ? 8 : 0);
        int d = kg - qg;
        bool ok = okc && (d >= -NW) && (d <= NW);
        sa[r] = ok ? (sa[r] + fm) : NEGF;
      }
    } else {                            // interior: fully in-band for all 16 rows
      float fm = (amrow[kg] != 0.f) ? NEGF : 0.f;
#pragma unroll
      for (int r = 0; r < 8; r++) sa[r] += fm;
    }
    s[n] = sa;
  }

  // ---- online softmax: row stats per lane-half, reduce over 16 lanes ----
#pragma unroll
  for (int r = 0; r < 8; r++) {
    float rm = fmaxf(s[0][r], s[1][r]);
    rm = fmaxf(rm, __shfl_xor(rm, 1, 32));
    rm = fmaxf(rm, __shfl_xor(rm, 2, 32));
    rm = fmaxf(rm, __shfl_xor(rm, 4, 32));
    rm = fmaxf(rm, __shfl_xor(rm, 8, 32));
    float mnew = fmaxf(m_run[r], rm);
    float corr = __expf(m_run[r] - mnew);     // ==0 once first real key appears
    float p0 = __expf(s[0][r] - mnew);        // masked cols underflow to 0
    float p1 = __expf(s[1][r] - mnew);
    float ps = p0 + p1;
    ps += __shfl_xor(ps, 1, 32);
    ps += __shfl_xor(ps, 2, 32);
    ps += __shfl_xor(ps, 4, 32);
    ps += __shfl_xor(ps, 8, 32);
    l_run[r] = l_run[r] * corr + ps;
    m_run[r] = mnew;
    o[0][r] *= corr; o[1][r] *= corr; o[2][r] *= corr; o[3][r] *= corr;
    int M = r + (hi ? 8 : 0);
    slab[M][l16]      = (h16)p0;              // C-layout -> LDS
    slab[M][16 + l16] = (h16)p1;
  }

  // ---- reload P as A-fragment (same wave; LDS ops in-order, no barrier) ----
  v16h Ap = cat8(*(const v8h*)(&slab[l16][a0]), *(const v8h*)(&slab[l16][a1]));

  // ---- P·V : V^T rows give contiguous 16-key B-fragment loads ----
  int kvb = k0 + b0off;
  int kvc = kvb < 0 ? 0 : (kvb > NS - 16 ? NS - 16 : kvb);
#pragma unroll
  for (int nt2 = 0; nt2 < 4; nt2++) {
    int dh = nt2 * 16 + l16;
    v16h Bv = *(const v16h*)(VTbh + (size_t)dh * NS + kvc);
    o[nt2] = wmma_f16(Ap, Bv, o[nt2]);
  }
}

// ---------------- banded flash attention: one wave = 16 queries ----------------
__global__ __launch_bounds__(256) void band_attn(
    const h16* __restrict__ Qh, const h16* __restrict__ Kh, const h16* __restrict__ VTh,
    const float* __restrict__ amask, const unsigned char* __restrict__ idxmask,
    float* __restrict__ out) {
  __shared__ __align__(32) h16 Pl[8][16][32];     // per-wave P slab (16x32 f16)

  const int wid  = threadIdx.x >> 5;
  const int lane = threadIdx.x & 31;
  const int l16  = lane & 15;
  const int hi   = lane >> 4;
  const int gw = blockIdx.x * 8 + wid;
  const int qt = gw & 255;                        // S/16 = 256 tiles per (b,h)
  const int bh = gw >> 8;
  const int b  = bh / NH;
  const int hh = bh % NH;
  const int q0 = qt * 16;

  const h16* Kbh  = Kh  + (((size_t)(b * NH + hh) * NS) << 6);
  const h16* VTbh = VTh + (size_t)(b * NH + hh) * DH * NS;
  const float* amrow = amask + (size_t)b * NS;

  const int a0 = hi ? 16 : 0, a1 = hi ? 24 : 8, b0off = hi ? 16 : 0;

  // Q A-fragments (Dh=64 -> two K=32 fragments), loaded once
  const h16* qrow = Qh + (((size_t)(b * NH + hh) * NS) << 6) + (size_t)(q0 + l16) * DH;
  v16h Aq0 = cat8(*(const v8h*)(qrow + a0),      *(const v8h*)(qrow + a1));
  v16h Aq1 = cat8(*(const v8h*)(qrow + 32 + a0), *(const v8h*)(qrow + 32 + a1));

  float m_run[8], l_run[8];
  v8f o[4] = {};
#pragma unroll
  for (int r = 0; r < 8; r++) { m_run[r] = NEGF; l_run[r] = 0.f; }

  for (int t = 0; t < 17; t++) {
    int k0 = q0 - NW + t * 32;
    bool edge = (t == 0) || (t == 16) || (k0 < 0) || (k0 + 32 > NS);
    // prefetch next step's K rows into cache (global_prefetch_b8)
    int kn = k0 + 32; kn = kn < 0 ? 0 : (kn > NS - 16 ? NS - 16 : kn);
    __builtin_prefetch((const void*)(Kbh + (size_t)(kn + l16) * DH), 0, 3);
    attn_step(k0, edge, q0, hi, l16, Kbh, VTbh, amrow,
              Aq0, Aq1, Pl[wid], a0, a1, b0off, m_run, l_run, o);
  }

  // ---- epilogue: normalize, apply is_index_masked, scatter f32 ----
#pragma unroll
  for (int r = 0; r < 8; r++) {
    int qg = q0 + r + (hi ? 8 : 0);
    bool im = idxmask[(size_t)b * NS + qg] != 0;
    float inv = (l_run[r] > 0.f && !im) ? 1.0f / l_run[r] : 0.f;
    float* orow = out + (size_t)(b * NS + qg) * ND + hh * DH;
    orow[ 0 + l16] = o[0][r] * inv;
    orow[16 + l16] = o[1][r] * inv;
    orow[32 + l16] = o[2][r] * inv;
    orow[48 + l16] = o[3][r] * inv;
  }
}

extern "C" void kernel_launch(void* const* d_in, const int* in_sizes, int n_in,
                              void* d_out, int out_size, void* d_ws, size_t ws_size,
                              hipStream_t stream) {
  (void)in_sizes; (void)n_in; (void)out_size; (void)ws_size;
  const float* X  = (const float*)d_in[0];
  const float* am = (const float*)d_in[1];
  const unsigned char* im = (const unsigned char*)d_in[2];
  const float* Wq = (const float*)d_in[3];
  const float* bq = (const float*)d_in[4];
  const float* Wk = (const float*)d_in[5];
  const float* bk = (const float*)d_in[6];
  const float* Wv = (const float*)d_in[7];
  const float* bv = (const float*)d_in[8];
  float* out = (float*)d_out;

  char* ws = (char*)d_ws;
  h16* Xh = (h16*)ws; ws += (size_t)BS * ND * 2;      // 12.6 MB
  h16* WT = (h16*)ws; ws += (size_t)3 * ND * ND * 2;  //  3.5 MB
  h16* Qh = (h16*)ws; ws += (size_t)BS * ND * 2;      // [B,H,S,64]
  h16* Kh = (h16*)ws; ws += (size_t)BS * ND * 2;      // [B,H,S,64]
  h16* VT = (h16*)ws; ws += (size_t)BS * ND * 2;      // [B,H,64,S]

  cvt_x<<<(BS * ND + 255) / 256, 256, 0, stream>>>(X, Xh, BS * ND);
  cvt_wT<<<(3 * ND * ND + 255) / 256, 256, 0, stream>>>(Wq, Wk, Wv, WT);

  dim3 gg((BS / 32) * (ND / 64), 3);                  // 3072 x 3 single-wave blocks
  qkv_gemm<<<gg, 32, 0, stream>>>(Xh, WT, bq, bk, bv, Qh, Kh, VT);

  band_attn<<<NB * NH * (NS / 16) / 8, 256, 0, stream>>>(Qh, Kh, VT, am, im, out);
}
